// PSR_GAT_82008105550064
// MI455X (gfx1250) — matrix operand order repaired and checked by reading, not compiled, not run on hardware
//
#include <hip/hip_runtime.h>
#include <hip/hip_bf16.h>
#include <math.h>

#define B_     256
#define N_     200
#define E_     800
#define NITEMS 50000
#define D_     64
#define K1_    180
#define K2_    162
#define STEPS_ 5
#define NPAD   208   // 13 * 16

typedef __attribute__((ext_vector_type(16))) _Float16 v16h;
typedef __attribute__((ext_vector_type(8)))  float    v8f;

__device__ __forceinline__ v8f wmma16(v16h a, v16h b, v8f c) {
  // D = A(16x32 f16) * B(32x16 f16) + C(16x16 f32)
  return __builtin_amdgcn_wmma_f32_16x16x32_f16(false, a, false, b, (short)0, c, false, false);
}

// A fragment (16x32) from LDS f16 matrix, row stride ld.
__device__ __forceinline__ v16h frag_a_lds(const _Float16* base, int ld, int m0, int k0, int lane) {
  int m = m0 + (lane & 15);
  int h = lane >> 4;
  const _Float16* r = base + m * ld + k0 + h * 8;
  v16h a;
#pragma unroll
  for (int j = 0; j < 8; ++j) a[j] = r[j];
#pragma unroll
  for (int j = 0; j < 8; ++j) a[8 + j] = r[16 + j];
  return a;
}

// A fragment (16x32) from LDS f32 matrix, row stride ld (convert to f16).
__device__ __forceinline__ v16h frag_a_f32(const float* base, int ld, int m0, int k0, int lane) {
  int m = m0 + (lane & 15);
  int h = lane >> 4;
  const float* r = base + m * ld + k0 + h * 8;
  v16h a;
#pragma unroll
  for (int j = 0; j < 8; ++j) a[j] = (_Float16)r[j];
#pragma unroll
  for (int j = 0; j < 8; ++j) a[8 + j] = (_Float16)r[16 + j];
  return a;
}

// B fragment (32x16) from global f32 row-major K x ldn matrix (B[k][n] = W[k*ldn + n]).
__device__ __forceinline__ v16h frag_b_f32(const float* W, int ldn, int k0, int n0, int lane) {
  int n = n0 + (lane & 15);
  int kb = (lane >> 4) * 8;
  const float* col = W + (size_t)(k0 + kb) * ldn + n;
  v16h b;
#pragma unroll
  for (int j = 0; j < 16; ++j) b[j] = (_Float16)col[(size_t)(j < 8 ? j : j + 8) * ldn];
  return b;
}

// B fragment (32x16) where B[k][n] = M[(n0+n)*ldk + k]  (M is N x K row-major -> transpose)
__device__ __forceinline__ v16h frag_b_trans(const float* M, int ldk, int k0, int n0, int lane) {
  int n = n0 + (lane & 15);
  int kb = k0 + ((lane >> 4) << 3);
  const float* r = M + (size_t)n * ldk + kb;
  v16h b;
#pragma unroll
  for (int j = 0; j < 16; ++j) b[j] = (_Float16)r[j < 8 ? j : j + 8];
  return b;
}

__device__ __forceinline__ float leakyf(float x, float s) { return x > 0.f ? x : s * x; }
__device__ __forceinline__ float sigmoidf_(float x) { return 1.f / (1.f + expf(-x)); }

__device__ __forceinline__ unsigned fenc(float x) {
  unsigned u = __float_as_uint(x);
  return (u & 0x80000000u) ? ~u : (u | 0x80000000u);
}
__device__ __forceinline__ float fdec(unsigned u) {
  unsigned v = (u & 0x80000000u) ? (u & 0x7fffffffu) : ~u;
  return __uint_as_float(v);
}

// ---------------------------------------------------------------------------
// GAT layer, fully LDS-resident. sx holds x (f16), shh is staging for h (f16).
// ---------------------------------------------------------------------------
static __device__ void gat_layer(_Float16* sx, _Float16* shh,
                                 const unsigned char* s_src, const unsigned char* s_dst,
                                 const unsigned char* s_msk, const float* s_ew, float* s_al,
                                 float* s_f, unsigned* s_u, float* s_av,
                                 const float* W, const float* a_s, const float* a_d,
                                 const float* bias, int nn, int lane, int wave, int tid)
{
  // preload attention vectors
  for (int i = tid; i < 2 * D_; i += 256) s_av[i] = (i < D_) ? a_s[i] : a_d[i - D_];
  // h = x @ W via WMMA (13 x 4 tiles across 8 waves)
  for (int t = wave; t < 13 * 4; t += 8) {
    int mt = (t >> 2) * 16, nt = (t & 3) * 16;
    v8f c = {};
#pragma unroll
    for (int kc = 0; kc < 2; ++kc)
      c = wmma16(frag_a_lds(sx, D_, mt, kc * 32, lane),
                 frag_b_f32(W, D_, kc * 32, nt, lane), c);
#pragma unroll
    for (int v = 0; v < 8; ++v) {
      int row = mt + v + ((lane >> 4) << 3);
      shh[row * D_ + nt + (lane & 15)] = (_Float16)c[v];
    }
  }
  // init segment max (encoded) / denominators
  for (int n = tid; n < NPAD; n += 256) { s_u[n] = 0u; s_f[n] = 0.f; }
  __syncthreads();

  // attention logits per edge + segment max
  for (int e = tid; e < E_; e += 256) {
    float lg = 0.f;
    if (s_msk[e]) {
      int s = s_src[e], d = s_dst[e];
      float acc = 0.f;
      for (int k = 0; k < D_; ++k)
        acc += (float)shh[s * D_ + k] * s_av[k] + (float)shh[d * D_ + k] * s_av[D_ + k];
      lg = leakyf(acc, 0.2f);
      atomicMax(&s_u[d], fenc(lg));
    }
    s_al[e] = lg;
  }
  __syncthreads();
  // decode max in place (empty segment -> 0, matching where(isfinite,mx,0))
  for (int n = tid; n < NPAD; n += 256) {
    float mx = (s_u[n] == 0u) ? 0.f : fdec(s_u[n]);
    s_u[n] = __float_as_uint(mx);
  }
  __syncthreads();
  // ex + denominator
  for (int e = tid; e < E_; e += 256) {
    float ex = 0.f;
    if (s_msk[e]) {
      int d = s_dst[e];
      ex = expf(s_al[e] - __uint_as_float(s_u[d]));
      atomicAdd(&s_f[d], ex);
    }
    s_al[e] = ex;
  }
  __syncthreads();
  // alpha = ex/den * ew
  for (int e = tid; e < E_; e += 256) {
    if (s_msk[e]) {
      int d = s_dst[e];
      s_al[e] = (s_al[e] / fmaxf(s_f[d], 1e-16f)) * s_ew[e];
    } else s_al[e] = 0.f;
  }
  __syncthreads();
  // zero x, then scatter-accumulate out = segsum(alpha * h[src])
  for (int i = tid; i < NPAD * D_; i += 256) sx[i] = (_Float16)0.f;
  __syncthreads();
  {
    int d = tid & 63;
    int chunk = tid >> 6;              // (dst & 3, d) ownership -> no write conflicts
    for (int e = 0; e < E_; ++e) {
      int dst = s_dst[e];
      if ((dst & 3) == chunk) {
        float a = s_al[e];
        if (a != 0.f) {
          int s = s_src[e];
          sx[dst * D_ + d] = (_Float16)((float)sx[dst * D_ + d] + a * (float)shh[s * D_ + d]);
        }
      }
    }
  }
  __syncthreads();
  // + bias, then leaky_relu (default slope 0.01)
  for (int i = tid; i < NPAD * D_; i += 256) {
    int row = i / D_, d = i % D_;
    float v = 0.f;
    if (row < nn) v = leakyf((float)sx[i] + bias[d], 0.01f);
    sx[i] = (_Float16)v;
  }
  __syncthreads();
}

// ---------------------------------------------------------------------------
// TopK pooling: stable descending rank == position in jax.lax.top_k order.
// ---------------------------------------------------------------------------
static __device__ void pool(_Float16* sx, _Float16* shh,
                            unsigned char* s_src, unsigned char* s_dst, unsigned char* s_msk,
                            float* s_ew, float* s_al, float* s_f, int* s_i, float* s_av,
                            float* s_scal, const float* p, int nin, int k, int tid)
{
  for (int i = tid; i < D_; i += 256) s_av[i] = p[i];
  if (tid == 0) {
    float nrm = 0.f;
    for (int d = 0; d < D_; ++d) nrm += p[d] * p[d];
    *s_scal = sqrtf(nrm);
  }
  __syncthreads();
  // scores
  for (int j = tid; j < nin; j += 256) {
    float acc = 0.f;
    for (int d = 0; d < D_; ++d) acc += (float)sx[j * D_ + d] * s_av[d];
    s_f[j] = tanhf(acc / *s_scal);
  }
  __syncthreads();
  // stable rank (ties -> lower index first)
  for (int j = tid; j < nin; j += 256) {
    float sj = s_f[j];
    int r = 0;
    for (int i2 = 0; i2 < nin; ++i2) {
      float si = s_f[i2];
      r += (si > sj) || (si == sj && i2 < j);
    }
    s_i[j] = r;
  }
  __syncthreads();
  // gather gated rows into staging
  for (int i = tid; i < NPAD * D_; i += 256) shh[i] = (_Float16)0.f;
  __syncthreads();
  for (int idx = tid; idx < nin * D_; idx += 256) {
    int j = idx / D_, d = idx % D_;
    int r = s_i[j];
    if (r < k) shh[r * D_ + d] = (_Float16)((float)sx[idx] * s_f[j]);
  }
  __syncthreads();
  for (int i = tid; i < NPAD * D_; i += 256) sx[i] = shh[i];
  __syncthreads();
  // remap edges; eattr = alpha gated by new mask
  for (int e = tid; e < E_; e += 256) {
    int ns = -1, nd = -1;
    if (s_msk[e]) {
      int rs = s_i[s_src[e]], rd = s_i[s_dst[e]];
      ns = (rs < k) ? rs : -1;
      nd = (rd < k) ? rd : -1;
    }
    bool m = (ns >= 0) && (nd >= 0);
    s_src[e] = m ? (unsigned char)ns : 0;
    s_dst[e] = m ? (unsigned char)nd : 0;
    s_ew[e] = m ? s_al[e] : 0.f;
    s_msk[e] = m ? 1 : 0;
  }
  __syncthreads();
}

static __device__ void dense_layer(_Float16* sx, _Float16* shh, const float* W,
                                   const float* bias, int nrows, int lane, int wave, int tid)
{
  for (int t = wave; t < 13 * 4; t += 8) {
    int mt = (t >> 2) * 16, nt = (t & 3) * 16;
    v8f c = {};
#pragma unroll
    for (int kc = 0; kc < 2; ++kc)
      c = wmma16(frag_a_lds(sx, D_, mt, kc * 32, lane),
                 frag_b_f32(W, D_, kc * 32, nt, lane), c);
#pragma unroll
    for (int v = 0; v < 8; ++v) {
      int row = mt + v + ((lane >> 4) << 3);
      shh[row * D_ + nt + (lane & 15)] = (_Float16)c[v];
    }
  }
  __syncthreads();
  for (int i = tid; i < NPAD * D_; i += 256) {
    int row = i / D_, d = i % D_;
    float v = 0.f;
    if (row < nrows) v = leakyf((float)shh[i] + bias[d], 0.01f);
    sx[i] = (_Float16)v;
  }
  __syncthreads();
}

// ---------------------------------------------------------------------------
// Kernel 1: row min/max of the six [B, NITEMS] score arrays.
// ---------------------------------------------------------------------------
__global__ __launch_bounds__(256) void k_minmax(
    const float* s0, const float* s1, const float* s2,
    const float* s3, const float* s4, const float* s5, float* mnmx)
{
  __shared__ float rmn[256], rmx[256];
  int b = blockIdx.x, a = blockIdx.y, tid = threadIdx.x;
  const float* sp[6] = {s0, s1, s2, s3, s4, s5};
  const float* s = sp[a] + (size_t)b * NITEMS;
  float mn = 3.4e38f, mx = -3.4e38f;
  for (int i = tid; i < NITEMS; i += 256) {
    float v = s[i];
    mn = fminf(mn, v); mx = fmaxf(mx, v);
  }
  rmn[tid] = mn; rmx[tid] = mx;
  __syncthreads();
  for (int o = 128; o > 0; o >>= 1) {
    if (tid < o) {
      rmn[tid] = fminf(rmn[tid], rmn[tid + o]);
      rmx[tid] = fmaxf(rmx[tid], rmx[tid + o]);
    }
    __syncthreads();
  }
  if (tid == 0) {
    mnmx[((size_t)a * B_ + b) * 2]     = rmn[0];
    mnmx[((size_t)a * B_ + b) * 2 + 1] = rmx[0];
  }
}

// ---------------------------------------------------------------------------
// Kernel 2: one workgroup per graph; full GAT->pool->GAT->pool->Wl1->Wl2.
// ---------------------------------------------------------------------------
__global__ __launch_bounds__(256) void k_graph(
    const int* xids, const int* eidx, const float* edur, const float* emb,
    const float* W1, const float* as1, const float* ad1, const float* b1, const float* p1,
    const float* W2, const float* as2, const float* ad2, const float* b2, const float* p2,
    const float* Wl1, const float* bl1, const float* Wl2, const float* bl2, float* Xout)
{
  __shared__ _Float16 sx[NPAD * D_];
  __shared__ _Float16 shh[NPAD * D_];
  __shared__ unsigned char s_src[E_], s_dst[E_], s_msk[E_];
  __shared__ float s_ew[E_];
  __shared__ float s_al[E_];
  __shared__ float s_f[NPAD];
  __shared__ unsigned s_u[NPAD];
  __shared__ float s_av[2 * D_];
  __shared__ float s_scal;

  const int b = blockIdx.x;
  const int tid = threadIdx.x;
  const int lane = tid & 31;
  const int wave = tid >> 5;

  // gather embeddings (rows >= N zero-padded)
  for (int idx = tid; idx < NPAD * D_; idx += 256) {
    int row = idx / D_, d = idx % D_;
    float v = 0.f;
    if (row < N_) v = emb[(size_t)xids[b * N_ + row] * D_ + d];
    sx[idx] = (_Float16)v;
  }
  // edges + duration weights
  for (int e = tid; e < E_; e += 256) {
    s_src[e] = (unsigned char)eidx[(size_t)b * 2 * E_ + e];
    s_dst[e] = (unsigned char)eidx[(size_t)b * 2 * E_ + E_ + e];
    s_msk[e] = 1;
    s_ew[e] = log1pf(edur[(size_t)b * E_ + e]);
  }
  __syncthreads();

  gat_layer(sx, shh, s_src, s_dst, s_msk, s_ew, s_al, s_f, s_u, s_av,
            W1, as1, ad1, b1, N_, lane, wave, tid);
  pool(sx, shh, s_src, s_dst, s_msk, s_ew, s_al, s_f, (int*)s_u, s_av, &s_scal,
       p1, N_, K1_, tid);
  gat_layer(sx, shh, s_src, s_dst, s_msk, s_ew, s_al, s_f, s_u, s_av,
            W2, as2, ad2, b2, K1_, lane, wave, tid);
  pool(sx, shh, s_src, s_dst, s_msk, s_ew, s_al, s_f, (int*)s_u, s_av, &s_scal,
       p2, K1_, K2_, tid);
  dense_layer(sx, shh, Wl1, bl1, K2_, lane, wave, tid);
  dense_layer(sx, shh, Wl2, bl2, K2_, lane, wave, tid);

  for (int idx = tid; idx < K2_ * D_; idx += 256)
    Xout[(size_t)b * K2_ * D_ + idx] = (float)sx[idx];
}

// ---------------------------------------------------------------------------
// Kernel 3: Set2Set; 16 graphs per block so the GRU mat-vecs become WMMA GEMMs.
// ---------------------------------------------------------------------------
__global__ __launch_bounds__(256) void k_set2set(
    const float* X, const float* Wih, const float* Whh,
    const float* bih, const float* bhh, float* qs_out)
{
  __shared__ float s_qs[16 * 128];
  __shared__ float s_h[16 * 64];
  __shared__ float s_gi[16 * 192];
  __shared__ float s_gh[16 * 192];
  __shared__ float s_a[16 * K2_];
  const int g0 = blockIdx.x * 16;
  const int tid = threadIdx.x, lane = tid & 31, wave = tid >> 5;

  for (int i = tid; i < 16 * 128; i += 256) s_qs[i] = 0.f;
  for (int i = tid; i < 16 * 64; i += 256) s_h[i] = 0.f;
  __syncthreads();

  for (int step = 0; step < STEPS_; ++step) {
    // gi = q_star @ Wih  (16x128 @ 128x192)
    for (int nt = wave; nt < 12; nt += 8) {
      v8f c = {};
#pragma unroll
      for (int kc = 0; kc < 4; ++kc)
        c = wmma16(frag_a_f32(s_qs, 128, 0, kc * 32, lane),
                   frag_b_f32(Wih, 192, kc * 32, nt * 16, lane), c);
#pragma unroll
      for (int v = 0; v < 8; ++v)
        s_gi[(v + ((lane >> 4) << 3)) * 192 + nt * 16 + (lane & 15)] = c[v];
    }
    // gh = h @ Whh  (16x64 @ 64x192)
    for (int nt = wave; nt < 12; nt += 8) {
      v8f c = {};
#pragma unroll
      for (int kc = 0; kc < 2; ++kc)
        c = wmma16(frag_a_f32(s_h, 64, 0, kc * 32, lane),
                   frag_b_f32(Whh, 192, kc * 32, nt * 16, lane), c);
#pragma unroll
      for (int v = 0; v < 8; ++v)
        s_gh[(v + ((lane >> 4) << 3)) * 192 + nt * 16 + (lane & 15)] = c[v];
    }
    __syncthreads();
    // GRU gates -> h (in place; each (g,j) independent)
    for (int idx = tid; idx < 16 * 64; idx += 256) {
      int g = idx >> 6, j = idx & 63;
      float ir = s_gi[g * 192 + j] + bih[j];
      float iz = s_gi[g * 192 + 64 + j] + bih[64 + j];
      float in = s_gi[g * 192 + 128 + j] + bih[128 + j];
      float hr = s_gh[g * 192 + j] + bhh[j];
      float hz = s_gh[g * 192 + 64 + j] + bhh[64 + j];
      float hn = s_gh[g * 192 + 128 + j] + bhh[128 + j];
      float r = sigmoidf_(ir + hr);
      float z = sigmoidf_(iz + hz);
      float ng = tanhf(in + r * hn);
      s_h[idx] = (1.f - z) * ng + z * s_h[idx];
    }
    __syncthreads();
    // attention logits over each graph's 162 nodes
    for (int idx = tid; idx < 16 * K2_; idx += 256) {
      int g = idx / K2_, n = idx % K2_;
      const float* xr = X + ((size_t)(g0 + g) * K2_ + n) * D_;
      float acc = 0.f;
      for (int d = 0; d < D_; ++d) acc += xr[d] * s_h[g * 64 + d];
      s_a[idx] = acc;
    }
    __syncthreads();
    // per-graph softmax + readout r; q_star = [h, r]
    for (int g = wave; g < 16; g += 8) {
      float mx = -1e30f;
      for (int n = lane; n < K2_; n += 32) mx = fmaxf(mx, s_a[g * K2_ + n]);
#pragma unroll
      for (int o = 16; o > 0; o >>= 1) mx = fmaxf(mx, __shfl_xor(mx, o, 32));
      float sum = 0.f;
      for (int n = lane; n < K2_; n += 32) {
        float e = expf(s_a[g * K2_ + n] - mx);
        s_a[g * K2_ + n] = e;
        sum += e;
      }
#pragma unroll
      for (int o = 16; o > 0; o >>= 1) sum += __shfl_xor(sum, o, 32);
      float inv = 1.f / sum;
      for (int d = lane; d < D_; d += 32) {
        float acc = 0.f;
        for (int n = 0; n < K2_; ++n)
          acc += s_a[g * K2_ + n] * X[((size_t)(g0 + g) * K2_ + n) * D_ + d];
        acc *= inv;
        s_qs[g * 128 + d] = s_h[g * 64 + d];
        s_qs[g * 128 + 64 + d] = acc;
      }
    }
    __syncthreads();
  }
  for (int i = tid; i < 16 * 128; i += 256)
    qs_out[(size_t)g0 * 128 + i] = s_qs[i];
}

// ---------------------------------------------------------------------------
// Kernel 4: g = [q_star, log1p(last2[:,0])] @ Wl3 + bl3     [B,129]@[129,64]
// ---------------------------------------------------------------------------
__global__ __launch_bounds__(64) void k_gvec(const float* qs, const float* last2,
                                             const float* Wl3, const float* bl3, float* g)
{
  int b = blockIdx.x, d = threadIdx.x;
  float acc = bl3[d];
  for (int k = 0; k < 128; ++k) acc += qs[b * 128 + k] * Wl3[k * 64 + d];
  acc += log1pf(last2[b * 2]) * Wl3[128 * 64 + d];
  g[b * 64 + d] = acc;
}

// ---------------------------------------------------------------------------
// Kernel 5: out = (g @ emb.T) * pref, fused streaming pref, WMMA GEMM.
// Block: 16 b-rows x 128 items (8 waves x one 16x16 tile each).
// ---------------------------------------------------------------------------
__global__ __launch_bounds__(256) void k_final(
    const float* g, const float* emb,
    const float* s0, const float* s1, const float* s2,
    const float* s3, const float* s4, const float* s5,
    const float* mnmx, const float* staw, float* out)
{
  __shared__ float s_g[16 * 64];
  __shared__ float s_mn[6 * 16], s_mx[6 * 16];
  __shared__ float s_w[6];
  const int b0 = blockIdx.y * 16;
  const int tid = threadIdx.x, lane = tid & 31, wave = tid >> 5;

  for (int i = tid; i < 16 * 64; i += 256) s_g[i] = g[(size_t)b0 * 64 + i];
  for (int i = tid; i < 6 * 16; i += 256) {
    int a = i / 16, bb = i % 16;
    s_mn[i] = mnmx[((size_t)a * B_ + b0 + bb) * 2];
    s_mx[i] = mnmx[((size_t)a * B_ + b0 + bb) * 2 + 1];
  }
  if (tid < 6) s_w[tid] = staw[tid];
  __syncthreads();

  const float* sp[6] = {s0, s1, s2, s3, s4, s5};
  int it0 = (blockIdx.x * 8 + wave) * 16;
  if (it0 >= NITEMS) return;

  // prefetch the next item tile's embedding rows while computing this one
  __builtin_prefetch(emb + (size_t)(it0 + 128) * D_, 0, 1);

  v8f c = {};
#pragma unroll
  for (int kc = 0; kc < 2; ++kc)
    c = wmma16(frag_a_f32(s_g, 64, 0, kc * 32, lane),
               frag_b_trans(emb, D_, kc * 32, it0, lane), c);

#pragma unroll
  for (int v = 0; v < 8; ++v) {
    int m = v + ((lane >> 4) << 3);
    int bidx = b0 + m;
    int item = it0 + (lane & 15);
    size_t off = (size_t)bidx * NITEMS + item;
    float pref = 0.f;
#pragma unroll
    for (int a = 0; a < 6; ++a) {
      float mn = s_mn[a * 16 + m], mxv = s_mx[a * 16 + m];
      float t = (sp[a][off] - mn) / (mxv - mn + 1e-12f);
      pref += s_w[a] * expf(t);
    }
    out[off] = c[v] * pref;
  }
}

// ---------------------------------------------------------------------------
extern "C" void kernel_launch(void* const* d_in, const int* in_sizes, int n_in,
                              void* d_out, int out_size, void* d_ws, size_t ws_size,
                              hipStream_t stream) {
  (void)in_sizes; (void)n_in; (void)out_size; (void)ws_size;
  const int*   xids  = (const int*)d_in[0];
  const int*   eidx  = (const int*)d_in[1];
  const float* edur  = (const float*)d_in[2];
  const float* last2 = (const float*)d_in[3];
  const float* sfm   = (const float*)d_in[4];
  const float* sdm   = (const float*)d_in[5];
  const float* sfw   = (const float*)d_in[6];
  const float* sdw   = (const float*)d_in[7];
  const float* sfd   = (const float*)d_in[8];
  const float* sdd   = (const float*)d_in[9];
  const float* emb   = (const float*)d_in[10];
  const float* W1    = (const float*)d_in[11];
  const float* as1   = (const float*)d_in[12];
  const float* ad1   = (const float*)d_in[13];
  const float* b1    = (const float*)d_in[14];
  const float* p1    = (const float*)d_in[15];
  const float* W2    = (const float*)d_in[16];
  const float* as2   = (const float*)d_in[17];
  const float* ad2   = (const float*)d_in[18];
  const float* b2    = (const float*)d_in[19];
  const float* p2    = (const float*)d_in[20];
  const float* Wl1   = (const float*)d_in[21];
  const float* bl1   = (const float*)d_in[22];
  const float* Wl2   = (const float*)d_in[23];
  const float* bl2   = (const float*)d_in[24];
  const float* Wih   = (const float*)d_in[25];
  const float* Whh   = (const float*)d_in[26];
  const float* bih   = (const float*)d_in[27];
  const float* bhh   = (const float*)d_in[28];
  const float* Wl3   = (const float*)d_in[29];
  const float* bl3   = (const float*)d_in[30];
  const float* staw  = (const float*)d_in[31];
  float* out = (float*)d_out;

  float* wsX = (float*)d_ws;                          // B * K2 * D
  float* wsQ = wsX + (size_t)B_ * K2_ * D_;           // B * 2D
  float* wsG = wsQ + (size_t)B_ * 2 * D_;             // B * D
  float* wsM = wsG + (size_t)B_ * D_;                 // 6 * B * 2

  k_minmax<<<dim3(B_, 6), 256, 0, stream>>>(sfm, sdm, sfw, sdw, sfd, sdd, wsM);
  k_graph<<<B_, 256, 0, stream>>>(xids, eidx, edur, emb,
                                  W1, as1, ad1, b1, p1,
                                  W2, as2, ad2, b2, p2,
                                  Wl1, bl1, Wl2, bl2, wsX);
  k_set2set<<<B_ / 16, 256, 0, stream>>>(wsX, Wih, Whh, bih, bhh, wsQ);
  k_gvec<<<B_, 64, 0, stream>>>(wsQ, last2, Wl3, bl3, wsG);
  k_final<<<dim3((NITEMS + 127) / 128, B_ / 16), 256, 0, stream>>>(
      wsG, emb, sfm, sdm, sfw, sdw, sfd, sdd, wsM, staw, out);
}
